// uniEncodeLSTM_52939766891156
// MI455X (gfx1250) — matrix operand order repaired and checked by reading, not compiled, not run on hardware
//
#include <hip/hip_runtime.h>

// ---------------------------------------------------------------------------
// 2-layer LSTM (B=64, T=1024, H=1024) for gfx1250 (MI455X).
// Persistent kernel, bf16 WMMA GEMMs with LDS-resident weights,
// layer pipelining across block groups, device-wide barrier per step.
// ---------------------------------------------------------------------------

#define BATCH   64
#define TLEN    1024
#define HDIM    1024
#define NBLK    128            // 64 layer-0 blocks + 64 layer-1 blocks
#define NTHR    256            // 8 wave32 waves

typedef __bf16 bf16_t;
typedef bf16_t v16bf __attribute__((ext_vector_type(16)));
typedef float  v8f   __attribute__((ext_vector_type(8)));
typedef unsigned int v4u __attribute__((ext_vector_type(4)));

union FragU { v4u u[2]; v16bf bf; };

// ---- dynamic LDS layout (bytes) -------------------------------------------
#define W_LDS_BYTES   (64 * 2048 * 2)            // 256 KB (layer1 worst case)
#define Z_OFF         (W_LDS_BYTES)
#define Z_BYTES       (64 * 64 * 4)              // 16 KB z staging (fp32)
#define BIAS_OFF      (Z_OFF + Z_BYTES)          // 64 floats
#define WI0_OFF       (BIAS_OFF + 256)           // 64 floats (layer0 only)
#define XS_OFF        (WI0_OFF + 256)            // 64 floats x column
#define SMEM_BYTES    (XS_OFF + 256)             // ~273 KB < 320 KB WGP LDS

// ---- workspace layout (bytes) ---------------------------------------------
#define WS_CNT_OFF    0
#define WS_GEN_OFF    4
#define WS_H0_OFF     256
#define HBUF_BYTES    (BATCH * HDIM * 2)         // 128 KB bf16
#define WS_H1_OFF     (WS_H0_OFF + 2 * HBUF_BYTES)
#define WS_ZERO_DW    ((WS_H1_OFF + 2 * HBUF_BYTES) / 4)

// ---------------------------------------------------------------------------
__device__ __forceinline__ unsigned short f2bf(float x) {
  unsigned u = __float_as_uint(x);
  u += 0x7FFFu + ((u >> 16) & 1u);               // round-to-nearest-even
  return (unsigned short)(u >> 16);
}
__device__ __forceinline__ float sigm(float x) {
  return 1.0f / (1.0f + __expf(-x));
}
__device__ __forceinline__ float tanh_fast(float x) {
  return 2.0f / (1.0f + __expf(-2.0f * x)) - 1.0f;
}

// A fragment (16x32 bf16, row-major source): per documented CDNA5 striping,
// lanes 0-15 hold k = [0..7] and [16..23]; lanes 16-31 hold k = [8..15],[24..31].
__device__ __forceinline__ v16bf ldA(const unsigned short* p, int laneHi) {
  FragU f;
  const v4u* q = (const v4u*)(p + (laneHi << 3));
  f.u[0] = q[0];      // 16B: k0..k0+7
  f.u[1] = q[2];      // 16B at +32B: k0+16..k0+23
  return f.bf;
}
// B fragment (32x16 bf16, column-major in LDS): lanes 0-15 hold k = [0..7],[16..23],
// lanes 16-31 hold k = [8..15],[24..31] (same striping as A).
__device__ __forceinline__ v16bf ldB(const unsigned short* p, int laneHi) {
  FragU f;
  const v4u* q = (const v4u*)(p + (laneHi << 3));
  f.u[0] = q[0];
  f.u[1] = q[2];
  return f.bf;
}

#define WMMA_BF16(a, b, c) \
  __builtin_amdgcn_wmma_f32_16x16x32_bf16(false, (a), false, (b), (short)0, (c), false, false)

// K=1024 contribution: 32 k-steps, 2 output tiles (two gates) per wave.
__device__ __forceinline__ void gemm1024(const unsigned short* __restrict__ arow,
                                         const unsigned short* __restrict__ w0,
                                         const unsigned short* __restrict__ w1,
                                         int laneHi, v8f& acc0, v8f& acc1) {
#pragma unroll 4
  for (int kk = 0; kk < 1024; kk += 32) {
    v16bf a  = ldA(arow + kk, laneHi);
    v16bf b0 = ldB(w0 + kk, laneHi);
    v16bf b1 = ldB(w1 + kk, laneHi);
    acc0 = WMMA_BF16(a, b0, acc0);
    acc1 = WMMA_BF16(a, b1, acc1);
  }
}

// Device-wide barrier: generation counter, agent-scope acquire/release.
__device__ __forceinline__ void grid_barrier(unsigned* cnt, unsigned* gen, unsigned target) {
  __syncthreads();
  if (threadIdx.x == 0) {
    __threadfence();
    unsigned prev = __hip_atomic_fetch_add(cnt, 1u, __ATOMIC_ACQ_REL, __HIP_MEMORY_SCOPE_AGENT);
    if (prev == NBLK - 1u) {
      __hip_atomic_store(cnt, 0u, __ATOMIC_RELAXED, __HIP_MEMORY_SCOPE_AGENT);
      __hip_atomic_store(gen, target, __ATOMIC_RELEASE, __HIP_MEMORY_SCOPE_AGENT);
    } else {
      while (__hip_atomic_load(gen, __ATOMIC_ACQUIRE, __HIP_MEMORY_SCOPE_AGENT) < target)
        __builtin_amdgcn_s_sleep(8);
    }
  }
  __syncthreads();
}

// ---------------------------------------------------------------------------
__global__ void init_ws_kernel(unsigned char* __restrict__ ws) {
  unsigned* p = (unsigned*)ws;
  for (unsigned i = blockIdx.x * blockDim.x + threadIdx.x; i < WS_ZERO_DW;
       i += gridDim.x * blockDim.x)
    p[i] = 0u;
}

// ---------------------------------------------------------------------------
__global__ void __launch_bounds__(NTHR, 1)
lstm2_persistent_kernel(const float* __restrict__ x,
                        const float* __restrict__ Wi0, const float* __restrict__ Wh0,
                        const float* __restrict__ b0,
                        const float* __restrict__ Wi1, const float* __restrict__ Wh1,
                        const float* __restrict__ b1,
                        float* __restrict__ out, unsigned char* __restrict__ ws) {
  extern __shared__ unsigned char smem[];
  unsigned short* wlds  = (unsigned short*)smem;
  float*          zbuf  = (float*)(smem + Z_OFF);
  float*          biass = (float*)(smem + BIAS_OFF);
  float*          wi0s  = (float*)(smem + WI0_OFF);
  float*          xs    = (float*)(smem + XS_OFF);

  const int tid  = threadIdx.x;
  const int bid  = blockIdx.x;
  const bool isL1 = (bid >= 64);
  const int  sub  = isL1 ? (bid - 64) : bid;   // column-group within the layer
  const int  n0   = sub << 4;                  // 16 gate-columns per block
  const int  K    = isL1 ? 2048 : 1024;

  unsigned* cnt = (unsigned*)(ws + WS_CNT_OFF);
  unsigned* gen = (unsigned*)(ws + WS_GEN_OFF);
  float* out_c  = out;
  float* out_h  = out + (size_t)BATCH * HDIM;
  float* out_h2 = out + (size_t)2 * BATCH * HDIM;

  // ---- one-time: convert this block's weight slice fp32 -> bf16 into LDS.
  // LDS layout: column-major, column c = gate*16 + j, contiguous in k, stride K.
  for (int e = tid; e < 64 * K; e += NTHR) {
    int j = e & 63;
    int k = e >> 6;
    int q = j >> 4, jj = j & 15;
    int gcol = q * HDIM + n0 + jj;             // column in 4H weight matrix
    float w;
    if (isL1) w = (k < HDIM) ? Wi1[(size_t)k * 4096 + gcol]
                             : Wh1[(size_t)(k - HDIM) * 4096 + gcol];
    else      w = Wh0[(size_t)k * 4096 + gcol];
    wlds[(size_t)j * K + k] = f2bf(w);
  }
  if (tid < 64) {
    int q = tid >> 4, jj = tid & 15;
    int gcol = q * HDIM + n0 + jj;
    biass[tid] = isL1 ? b1[gcol] : b0[gcol];
    if (!isL1) wi0s[tid] = Wi0[gcol];
  }
  __syncthreads();

  // ---- wave / lane geometry -----------------------------------------------
  const int lane   = tid & 31;
  const int wave   = tid >> 5;
  const int mi     = wave & 3;                 // M tile (rows 16*mi..)
  const int gp     = wave >> 2;                // gate pair: gates {2gp, 2gp+1}
  const int laneHi = (lane >> 4) & 1;
  const int n      = lane & 15;
  const int row    = (mi << 4) + n;            // A-matrix row for this lane

  const unsigned short* w0col = wlds + (size_t)((gp << 5) + n) * K;        // gate 2gp
  const unsigned short* w1col = w0col + (size_t)16 * K;                    // gate 2gp+1

  float cst[4] = {0.f, 0.f, 0.f, 0.f};         // persistent cell state (4 elems/thread)

  // ---- pipelined time loop: layer0 does t=k, layer1 does t=k-1 ------------
  for (int k = 0; k <= TLEN; ++k) {
    const bool active = isL1 ? (k >= 1) : (k < TLEN);
    if (active) {
      const int t = isL1 ? (k - 1) : k;
      const unsigned short* h0r =
          (const unsigned short*)(ws + WS_H0_OFF + (size_t)((k + 1) & 1) * HBUF_BYTES);

      if (!isL1 && tid < BATCH) xs[tid] = x[tid * TLEN + t];

      v8f acc0 = {0.f, 0.f, 0.f, 0.f, 0.f, 0.f, 0.f, 0.f};
      v8f acc1 = {0.f, 0.f, 0.f, 0.f, 0.f, 0.f, 0.f, 0.f};

      gemm1024(h0r + row * HDIM, w0col, w1col, laneHi, acc0, acc1);
      if (isL1) {
        const unsigned short* h1r =
            (const unsigned short*)(ws + WS_H1_OFF + (size_t)(k & 1) * HBUF_BYTES);
        gemm1024(h1r + row * HDIM, w0col + 1024, w1col + 1024, laneHi, acc0, acc1);
      }

      // stage z tiles to LDS (C/D layout: VGPR r -> row 16*mi + r + 8*laneHi)
#pragma unroll
      for (int r = 0; r < 8; ++r) {
        int m = (mi << 4) + r + (laneHi << 3);
        zbuf[m * 64 + (gp << 5) + n]      = acc0[r];
        zbuf[m * 64 + (gp << 5) + 16 + n] = acc1[r];
      }
      __syncthreads();

      // element-wise gates: each thread owns 4 (batch-row, col) cells
      unsigned short* h0w = (unsigned short*)(ws + WS_H0_OFF + (size_t)(k & 1) * HBUF_BYTES);
      unsigned short* h1w = (unsigned short*)(ws + WS_H1_OFF + (size_t)((k + 1) & 1) * HBUF_BYTES);
#pragma unroll
      for (int i = 0; i < 4; ++i) {
        int e = tid + NTHR * i;
        int m = e >> 4, j = e & 15;
        float zi = zbuf[m * 64 + j]      + biass[j];
        float zf = zbuf[m * 64 + 16 + j] + biass[16 + j];
        float zg = zbuf[m * 64 + 32 + j] + biass[32 + j];
        float zo = zbuf[m * 64 + 48 + j] + biass[48 + j];
        if (!isL1) {
          float xv = xs[m];
          zi += xv * wi0s[j];
          zf += xv * wi0s[16 + j];
          zg += xv * wi0s[32 + j];
          zo += xv * wi0s[48 + j];
        }
        float c = sigm(zf) * cst[i] + sigm(zi) * tanh_fast(zg);
        float h = sigm(zo) * tanh_fast(c);
        cst[i] = c;
        if (isL1) {
          out_h2[(size_t)m * (TLEN * HDIM) + (size_t)t * HDIM + n0 + j] = h;
          h1w[m * HDIM + n0 + j] = f2bf(h);
          if (t == TLEN - 1) {
            out_c[m * HDIM + n0 + j] = c;
            out_h[m * HDIM + n0 + j] = h;
          }
        } else {
          h0w[m * HDIM + n0 + j] = f2bf(h);
        }
      }
    }
    if (k < TLEN) grid_barrier(cnt, gen, (unsigned)(k + 1));
  }
}

// ---------------------------------------------------------------------------
extern "C" void kernel_launch(void* const* d_in, const int* in_sizes, int n_in,
                              void* d_out, int out_size, void* d_ws, size_t ws_size,
                              hipStream_t stream) {
  const float* x   = (const float*)d_in[0];
  const float* Wi0 = (const float*)d_in[1];
  const float* Wh0 = (const float*)d_in[2];
  const float* b0  = (const float*)d_in[3];
  const float* Wi1 = (const float*)d_in[4];
  const float* Wh1 = (const float*)d_in[5];
  const float* b1  = (const float*)d_in[6];
  (void)in_sizes; (void)n_in; (void)out_size; (void)ws_size;

  // Allow ~273 KB dynamic LDS (WGP supports up to 320 KB). Host-side, capture-safe.
  (void)hipFuncSetAttribute((const void*)lstm2_persistent_kernel,
                            hipFuncAttributeMaxDynamicSharedMemorySize, SMEM_BYTES);

  hipLaunchKernelGGL(init_ws_kernel, dim3(128), dim3(256), 0, stream,
                     (unsigned char*)d_ws);
  hipLaunchKernelGGL(lstm2_persistent_kernel, dim3(NBLK), dim3(NTHR), SMEM_BYTES, stream,
                     x, Wi0, Wh0, b0, Wi1, Wh1, b1,
                     (float*)d_out, (unsigned char*)d_ws);
}
